// CustomTorchModelCfc_61220463837853
// MI455X (gfx1250) — matrix operand order repaired and checked by reading, not compiled, not run on hardware
//
#include <hip/hip_runtime.h>
#include <hip/hip_bf16.h>

typedef __attribute__((ext_vector_type(16))) _Float16 v16h;
typedef __attribute__((ext_vector_type(8)))  _Float16 v8h;
typedef __attribute__((ext_vector_type(4)))  _Float16 v4h;
typedef __attribute__((ext_vector_type(8)))  float    v8f;

#define NB    1024   // batch
#define TT    256    // timesteps
#define NOBS  64     // obs features
#define WPB   2      // waves per block, each wave owns a 16-row batch tile
#define NTHR  (WPB * 32)

__device__ __forceinline__ void wait_lds() {
  asm volatile("s_wait_dscnt 0" ::: "memory");
}

// Branch-free tanh: native v_tanh_f32 on gfx1250 if available, else exp2+rcp.
__device__ __forceinline__ float fast_tanh(float x) {
#if __has_builtin(__builtin_amdgcn_tanhf)
  return __builtin_amdgcn_tanhf(x);
#else
  float ax = __builtin_fabsf(x);
  float e = __builtin_amdgcn_exp2f(ax * -2.8853900817779268f);  // e^(-2ax)
  float r = 1.0f - 2.0f * e * __builtin_amdgcn_rcpf(1.0f + e);
  return __builtin_copysignf(r, x);
#endif
}
__device__ __forceinline__ float fast_sigmoid(float x) {
  return __builtin_fmaf(0.5f, fast_tanh(0.5f * x), 0.5f);
}

__device__ __forceinline__ v8f wmma16(v16h a, v16h b, v8f c) {
  return __builtin_amdgcn_wmma_f32_16x16x32_f16(false, a, false, b, (short)0, c,
                                                false, false);
}

// A fragment: 16x32 f16; lane%16 = M row; lanes<16 hold K={0..7,16..23},
// lanes>=16 hold K={8..15,24..31}.
__device__ __forceinline__ v16h frag_a(const _Float16* p, int S, int row16,
                                       int hi, int kc) {
  const int off = kc * 32 + hi * 8;
  union { v16h v; v8h h[2]; } u;
  u.h[0] = *(const v8h*)(p + row16 * S + off);
  u.h[1] = *(const v8h*)(p + row16 * S + off + 16);
  return u.v;
}

// B fragment: 32x16 f16; lane%16 = N col; lanes<16 hold K={0..7,16..23},
// lanes>=16 K={8..15,24..31}. Weights stored in LDS as [n][k] row-major.
__device__ __forceinline__ v16h frag_b(const _Float16* p, int S, int row16,
                                       int hi, int nt, int kc) {
  const int base = (nt * 16 + row16) * S + kc * 32 + hi * 8;
  union { v16h v; v8h h[2]; } u;
  u.h[0] = *(const v8h*)(p + base);
  u.h[1] = *(const v8h*)(p + base + 16);
  return u.v;
}

template <int KC, int NTL>
__device__ __forceinline__ void load_bfrags(const _Float16* w, int wS,
                                            int row16, int hi, v16h* dst) {
#pragma unroll
  for (int nt = 0; nt < NTL; ++nt)
#pragma unroll
    for (int kc = 0; kc < KC; ++kc)
      dst[nt * KC + kc] = frag_b(w, wS, row16, hi, nt, kc);
}

// One CfC layer for a 16-batch tile. Gate weights pre-summed (ta+tb) so only
// 3 WMMA chains. B-fragments and biases live in registers (loop-invariant).
template <int KC, int NTL, int H>
__device__ __forceinline__ void cfc_layer(
    const _Float16* inb, int inS,
    const v16h* Bw1, const v16h* Bw2, const v16h* Bwt,
    const float* bb1, const float* bb2, const float* bbt,  // [NTL] in regs
    _Float16* dst1, int d1S, _Float16* dst2, int d2S, bool relu2,
    int row16, int hi) {
  wait_lds();
  v16h A[KC];
#pragma unroll
  for (int kc = 0; kc < KC; ++kc) A[kc] = frag_a(inb, inS, row16, hi, kc);
#pragma unroll
  for (int nt = 0; nt < NTL; ++nt) {
    v8f a1 = {}, a2 = {}, at = {};
#pragma unroll
    for (int kc = 0; kc < KC; ++kc) {
      a1 = wmma16(A[kc], Bw1[nt * KC + kc], a1);
      a2 = wmma16(A[kc], Bw2[nt * KC + kc], a2);
      at = wmma16(A[kc], Bwt[nt * KC + kc], at);
    }
    const int n = nt * 16 + row16;
#pragma unroll
    for (int r = 0; r < 8; ++r) {
      const int m = hi ? (r + 8) : r;  // C/D layout: vgpr r -> M=r / r+8
      float tg = fast_sigmoid(at[r] + bbt[nt]);
      float f1 = fast_tanh(a1[r] + bb1[nt]);
      float f2 = fast_tanh(a2[r] + bb2[nt]);
      float o  = __builtin_fmaf(tg, f2 - f1, f1);
      if (n < H) {
        dst1[m * d1S + n] = (_Float16)o;
        dst2[m * d2S + n] = relu2 ? (_Float16)fmaxf(o, 0.0f) : (_Float16)o;
      }
    }
  }
}

__global__ __launch_bounds__(NTHR, 1) void cfc_rnn_kernel(
    const float* __restrict__ x, const float* __restrict__ h0,
    const float* __restrict__ w1_0, const float* __restrict__ w2_0,
    const float* __restrict__ taw_0, const float* __restrict__ tbw_0,
    const float* __restrict__ b1_0, const float* __restrict__ b2_0,
    const float* __restrict__ tab_0, const float* __restrict__ tbb_0,
    const float* __restrict__ mask_0,
    const float* __restrict__ w1_1, const float* __restrict__ w2_1,
    const float* __restrict__ taw_1, const float* __restrict__ tbw_1,
    const float* __restrict__ b1_1, const float* __restrict__ b2_1,
    const float* __restrict__ tab_1, const float* __restrict__ tbb_1,
    const float* __restrict__ mask_1,
    const float* __restrict__ w1_2, const float* __restrict__ w2_2,
    const float* __restrict__ taw_2, const float* __restrict__ tbw_2,
    const float* __restrict__ b1_2, const float* __restrict__ b2_2,
    const float* __restrict__ tab_2, const float* __restrict__ tbb_2,
    const float* __restrict__ mask_2,
    const float* __restrict__ aw, const float* __restrict__ ab,
    const float* __restrict__ vw, const float* __restrict__ vb,
    float* __restrict__ out) {
  // Weights, f16, pre-masked / pre-summed, B-fragment layout [n][k], K padded.
  __shared__ __align__(32) _Float16 sW0[3][32 * 96];  // L0: h20->32, c84->96
  __shared__ __align__(32) _Float16 sW1[3][16 * 32];  // L1: h12->16, c32
  __shared__ __align__(32) _Float16 sW2[3][32 * 64];  // L2: h32,     c44->64
  __shared__ __align__(32) _Float16 sWH[16 * 32];     // head: n0-7 action, n8 value
  __shared__ float sB0[3][32], sB1[3][16], sB2[3][32], sBH[16];
  // Per-wave recurrent/input buffers: [row][k] f16, pads stay zero.
  __shared__ __align__(32) _Float16 sIn0[WPB][16 * 96];  // x(64)|h_a(20)|pad
  __shared__ __align__(32) _Float16 sIn1[WPB][16 * 32];  // n_a(20)|h_b(12)
  __shared__ __align__(32) _Float16 sIn2[WPB][16 * 64];  // n_b(12)|h_c(32)|pad
  __shared__ __align__(32) _Float16 sY[WPB][16 * 32];    // relu(n_c)

  const int tid = threadIdx.x;

  // ---- zero all LDS (pads must be zero) ----
  for (int i = tid; i < 3 * 32 * 96; i += NTHR) (&sW0[0][0])[i] = (_Float16)0.f;
  for (int i = tid; i < 3 * 16 * 32; i += NTHR) (&sW1[0][0])[i] = (_Float16)0.f;
  for (int i = tid; i < 3 * 32 * 64; i += NTHR) (&sW2[0][0])[i] = (_Float16)0.f;
  for (int i = tid; i < 16 * 32; i += NTHR)     sWH[i] = (_Float16)0.f;
  for (int i = tid; i < 3 * 32; i += NTHR) (&sB0[0][0])[i] = 0.f;
  for (int i = tid; i < 3 * 16; i += NTHR) (&sB1[0][0])[i] = 0.f;
  for (int i = tid; i < 3 * 32; i += NTHR) (&sB2[0][0])[i] = 0.f;
  for (int i = tid; i < 16; i += NTHR)     sBH[i] = 0.f;
  for (int i = tid; i < WPB * 16 * 96; i += NTHR) (&sIn0[0][0])[i] = (_Float16)0.f;
  for (int i = tid; i < WPB * 16 * 32; i += NTHR) (&sIn1[0][0])[i] = (_Float16)0.f;
  for (int i = tid; i < WPB * 16 * 64; i += NTHR) (&sIn2[0][0])[i] = (_Float16)0.f;
  for (int i = tid; i < WPB * 16 * 32; i += NTHR) (&sY[0][0])[i] = (_Float16)0.f;
  __syncthreads();

  // ---- stage weights: mask w1/w2, sum gate weights ta+tb ----
  auto stageM = [&](const float* w, const float* mk, _Float16* d, int h, int c, int S) {
    for (int i = tid; i < h * c; i += NTHR) {
      int n = i / c, k = i - n * c;
      d[n * S + k] = (_Float16)(w[i] * mk[i]);
    }
  };
  auto stageS = [&](const float* wa, const float* wb, _Float16* d, int h, int c, int S) {
    for (int i = tid; i < h * c; i += NTHR) {
      int n = i / c, k = i - n * c;
      d[n * S + k] = (_Float16)(wa[i] + wb[i]);
    }
  };
  stageM(w1_0, mask_0, sW0[0], 20, 84, 96);
  stageM(w2_0, mask_0, sW0[1], 20, 84, 96);
  stageS(taw_0, tbw_0, sW0[2], 20, 84, 96);
  stageM(w1_1, mask_1, sW1[0], 12, 32, 32);
  stageM(w2_1, mask_1, sW1[1], 12, 32, 32);
  stageS(taw_1, tbw_1, sW1[2], 12, 32, 32);
  stageM(w1_2, mask_2, sW2[0], 32, 44, 64);
  stageM(w2_2, mask_2, sW2[1], 32, 44, 64);
  stageS(taw_2, tbw_2, sW2[2], 32, 44, 64);
  for (int i = tid; i < 8 * 32; i += NTHR) {
    int n = i >> 5, k = i & 31;
    sWH[n * 32 + k] = (_Float16)aw[i];
  }
  for (int i = tid; i < 32; i += NTHR) sWH[8 * 32 + i] = (_Float16)vw[i];
  for (int i = tid; i < 20; i += NTHR) {
    sB0[0][i] = b1_0[i]; sB0[1][i] = b2_0[i]; sB0[2][i] = tab_0[i] + tbb_0[i];
  }
  for (int i = tid; i < 12; i += NTHR) {
    sB1[0][i] = b1_1[i]; sB1[1][i] = b2_1[i]; sB1[2][i] = tab_1[i] + tbb_1[i];
  }
  for (int i = tid; i < 32; i += NTHR) {
    sB2[0][i] = b1_2[i]; sB2[1][i] = b2_2[i]; sB2[2][i] = tab_2[i] + tbb_2[i];
  }
  for (int i = tid; i < 8; i += NTHR) sBH[i] = ab[i];
  if (tid == 0) sBH[8] = vb[0];

  // ---- stage h0 into recurrent slots ----
  for (int i = tid; i < WPB * 16 * 64; i += NTHR) {
    int w = i >> 10, r = (i >> 6) & 15, c = i & 63;
    int b = blockIdx.x * (WPB * 16) + w * 16 + r;
    float v = h0[b * 64 + c];
    if (c < 20)      sIn0[w][r * 96 + 64 + c]        = (_Float16)v;
    else if (c < 32) sIn1[w][r * 32 + 20 + (c - 20)] = (_Float16)v;
    else             sIn2[w][r * 64 + 12 + (c - 32)] = (_Float16)v;
  }
  __syncthreads();

  // ---- per-wave setup ----
  const int lane = tid & 31;
  const int wv = tid >> 5;
  const int row16 = lane & 15, hi = lane >> 4;
  const int b0 = blockIdx.x * (WPB * 16) + wv * 16;
  _Float16* in0 = sIn0[wv];
  _Float16* in1 = sIn1[wv];
  _Float16* in2 = sIn2[wv];
  _Float16* yb  = sY[wv];
  const size_t VOFF  = (size_t)NB * TT * 8;
  const size_t HXOFF = VOFF + (size_t)NB * TT;
  const float* xrow = x + (size_t)(b0 + row16) * TT * NOBS + hi * 32;

  // Hoist ALL loop-invariant weight fragments + biases into registers.
  v16h B0w1[6], B0w2[6], B0wt[6];
  load_bfrags<3, 2>(sW0[0], 96, row16, hi, B0w1);
  load_bfrags<3, 2>(sW0[1], 96, row16, hi, B0w2);
  load_bfrags<3, 2>(sW0[2], 96, row16, hi, B0wt);
  v16h B1w1[1], B1w2[1], B1wt[1];
  load_bfrags<1, 1>(sW1[0], 32, row16, hi, B1w1);
  load_bfrags<1, 1>(sW1[1], 32, row16, hi, B1w2);
  load_bfrags<1, 1>(sW1[2], 32, row16, hi, B1wt);
  v16h B2w1[4], B2w2[4], B2wt[4];
  load_bfrags<2, 2>(sW2[0], 64, row16, hi, B2w1);
  load_bfrags<2, 2>(sW2[1], 64, row16, hi, B2w2);
  load_bfrags<2, 2>(sW2[2], 64, row16, hi, B2wt);
  v16h BH = frag_b(sWH, 32, row16, hi, 0, 0);

  float b1_0v[2], b2_0v[2], bt_0v[2], b1_2v[2], b2_2v[2], bt_2v[2];
#pragma unroll
  for (int nt = 0; nt < 2; ++nt) {
    const int n = nt * 16 + row16;
    b1_0v[nt] = sB0[0][n]; b2_0v[nt] = sB0[1][n]; bt_0v[nt] = sB0[2][n];
    b1_2v[nt] = sB2[0][n]; b2_2v[nt] = sB2[1][n]; bt_2v[nt] = sB2[2][n];
  }
  float b1_1v[1] = {sB1[0][row16]};
  float b2_1v[1] = {sB1[1][row16]};
  float bt_1v[1] = {sB1[2][row16]};
  const float bbH = sBH[row16];

  // ---- main scan: one wave == one 16-batch tile, fully LDS/reg-resident ----
#pragma unroll 1
  for (int t = 0; t < TT; ++t) {
    // stage x_t tile: each lane converts 32 floats (half a row) to f16 in LDS
    const float4* xp = (const float4*)(xrow + (size_t)t * NOBS);
#pragma unroll
    for (int j = 0; j < 8; ++j) {
      float4 f = xp[j];
      v4h hv;
      hv.x = (_Float16)f.x; hv.y = (_Float16)f.y;
      hv.z = (_Float16)f.z; hv.w = (_Float16)f.w;
      *(v4h*)(in0 + row16 * 96 + hi * 32 + 4 * j) = hv;
    }
    if (t + 1 < TT)
      __builtin_prefetch((const char*)xp + NOBS * sizeof(float), 0, 3);

    cfc_layer<3, 2, 20>(in0, 96, B0w1, B0w2, B0wt, b1_0v, b2_0v, bt_0v,
                        in0 + 64, 96, in1, 32, false, row16, hi);
    cfc_layer<1, 1, 12>(in1, 32, B1w1, B1w2, B1wt, b1_1v, b2_1v, bt_1v,
                        in1 + 20, 32, in2, 64, false, row16, hi);
    cfc_layer<2, 2, 32>(in2, 64, B2w1, B2w2, B2wt, b1_2v, b2_2v, bt_2v,
                        in2 + 12, 64, yb, 32, true, row16, hi);

    // fused heads: y(16x32) x [action|value](32x16) in one WMMA
    wait_lds();
    v16h Ah = frag_a(yb, 32, row16, hi, 0);
    v8f ch = {};
    ch = wmma16(Ah, BH, ch);
#pragma unroll
    for (int r = 0; r < 8; ++r) {
      const int m = hi ? (r + 8) : r;
      const size_t b = (size_t)(b0 + m);
      const float v = ch[r] + bbH;
      if (row16 < 8)       out[(b * TT + t) * 8 + row16] = v;
      else if (row16 == 8) out[VOFF + b * TT + t] = v;
    }
  }

  // ---- final hidden state hx = concat(h_a, h_b, h_c) ----
  wait_lds();
  {
    const int r = row16, b = b0 + r;
#pragma unroll 1
    for (int j = 0; j < 32; ++j) {
      const int c = hi * 32 + j;
      float v;
      if (c < 20)      v = (float)in0[r * 96 + 64 + c];
      else if (c < 32) v = (float)in1[r * 32 + 20 + (c - 20)];
      else             v = (float)in2[r * 64 + 12 + (c - 32)];
      out[HXOFF + (size_t)b * 64 + c] = v;
    }
  }
}

extern "C" void kernel_launch(void* const* d_in, const int* in_sizes, int n_in,
                              void* d_out, int out_size, void* d_ws,
                              size_t ws_size, hipStream_t stream) {
  (void)in_sizes; (void)n_in; (void)out_size; (void)d_ws; (void)ws_size;
  const float* p[33];
  for (int i = 0; i < 33; ++i) p[i] = (const float*)d_in[i];
  dim3 grid(NB / (WPB * 16));
  dim3 block(NTHR);
  cfc_rnn_kernel<<<grid, block, 0, stream>>>(
      p[0], p[1],
      p[2], p[3], p[4], p[5], p[6], p[7], p[8], p[9], p[10],
      p[11], p[12], p[13], p[14], p[15], p[16], p[17], p[18], p[19],
      p[20], p[21], p[22], p[23], p[24], p[25], p[26], p[27], p[28],
      p[29], p[30], p[31], p[32],
      (float*)d_out);
}
